// Codebook_42219528519948
// MI455X (gfx1250) — compile-verified, matrix operand order checked by reading, not run on hardware
//
#include <hip/hip_runtime.h>

// ---------------------------------------------------------------------------
// VQ-VAE codebook: fused distance-GEMM (bf16 WMMA) + argmin, then gather+loss.
//   z:[2,256,16,56,56] f32, key/value:[1024,256] f32
//   N = 100352 rows, C = 256, K = 1024  ->  2*N*C*K = 52.6 GFLOP matmul
// HBM ~300MB (~13us @23.3TB/s); WMMA bf16 16x16x32 is the matrix path.
// Each wave owns 32 rows (2 A sub-tiles) so every B fragment from LDS feeds
// two independent v_wmma chains (hides ds latency, halves B LDS traffic).
// ---------------------------------------------------------------------------

#define CDIM  256
#define KCODES 1024
#define SVOL  50176      // D*H*W = 16*56*56
#define NROWS 100352     // B*SVOL
#define BETA  0.25f

typedef __attribute__((ext_vector_type(16))) __bf16 v16bf;
typedef __attribute__((ext_vector_type(8)))  float  v8f;

__device__ __forceinline__ unsigned short f2bf(float f) {
    unsigned int u = __builtin_bit_cast(unsigned int, f);
    unsigned int r = u + 0x7FFFu + ((u >> 16) & 1u);   // round-to-nearest-even
    return (unsigned short)(r >> 16);
}

// ISA 7.12.2: 16-bit A (16x32) fragment element (m,k) -> (lane, half) mapping.
// lanes 0-15 hold K in {0..7, 16..23}; lanes 16-31 hold K in {8..15, 24..31}.
// B (32x16) assumed symmetric with n in place of m.
__device__ __forceinline__ int fragLane(int m, int k) { return m + (((k >> 3) & 1) << 4); }
__device__ __forceinline__ int fragHalf(int k)        { return (k & 7) + ((k >> 4) << 3); }

// ---------------------------------------------------------------------------
// Kernel 0: codebook -> bf16 B-fragment layout in ws, code norms, zero loss.
// Fragment block (t = code/16, q = c/32): 32 lanes x 16 halves (1KB), blocks
// ordered (t,q) so a 128-code chunk is one contiguous 64KB slab.
// ---------------------------------------------------------------------------
__global__ void vq_prep(const float* __restrict__ kw,
                        unsigned short* __restrict__ wsCb,
                        float* __restrict__ wsNorm,
                        float* __restrict__ wsLoss) {
    const int k = blockIdx.x;      // code id
    const int c = threadIdx.x;     // channel
    if (k == 0 && c == 0) *wsLoss = 0.0f;

    float v = kw[k * CDIM + c];

    __shared__ float red[256];
    red[c] = v * v;
    __syncthreads();
    for (int s = 128; s > 0; s >>= 1) {
        if (c < s) red[c] += red[c + s];
        __syncthreads();
    }
    if (c == 0) wsNorm[k] = red[0];

    const int t = k >> 4, n0 = k & 15, q = c >> 5, kk = c & 31;
    wsCb[(((t * 8 + q) * 32) + fragLane(n0, kk)) * 16 + fragHalf(kk)] = f2bf(v);
}

// ---------------------------------------------------------------------------
// Kernel 1: fused ||e||^2 - 2 z.e^T + running argmin.
// 784 WGs x 128 threads (4 waves). WG tile: 128 rows x all 1024 codes.
// LDS: A tile 64KB (fragment layout) + B tile 64KB (copied pre-formatted).
// Wave owns 32 rows: aFrag0/aFrag1 (2x64 VGPRs) reused across all 64 code
// sub-tiles; each B fragment feeds two independent WMMA accumulator chains.
// ---------------------------------------------------------------------------
__global__ void __launch_bounds__(128)
vq_argmin(const float* __restrict__ z,
          const unsigned short* __restrict__ wsCb,
          const float* __restrict__ wsNorm,
          int* __restrict__ wsIdx,
          float* __restrict__ outIdxF) {
    extern __shared__ __align__(32) unsigned short lds[];
    unsigned short* aT = lds;            // 32768 ushorts = 64KB
    unsigned short* bT = lds + 32768;    // 32768 ushorts = 64KB

    const int tid  = threadIdx.x;        // 0..127
    const int lane = tid & 31;
    const int wave = tid >> 5;           // 0..3
    const int rowBase = blockIdx.x * 128;
    const int b  = rowBase / SVOL;       // tile never spans batch (50176%128==0)
    const int s0 = rowBase - b * SVOL;

    // --- stage z tile into A-fragment layout: b128 loads over rows -------
    {
        const int r4 = (tid & 31) * 4;                 // row group base, fixed/thread
        for (int i = 0; i < 64; ++i) {
            const int c = (tid >> 5) + i * 4;          // 0..255
            const int q = c >> 5, kk = c & 31;
            const float4 v = *(const float4*)(
                z + ((size_t)(b * CDIM + c)) * SVOL + (size_t)(s0 + r4));
            const int h = fragHalf(kk);
#pragma unroll
            for (int u = 0; u < 4; ++u) {
                const int r = r4 + u;
                const float vv = (u == 0) ? v.x : (u == 1) ? v.y : (u == 2) ? v.z : v.w;
                aT[((((r >> 4) * 8 + q) * 32) + fragLane(r & 15, kk)) * 16 + h] = f2bf(vv);
            }
        }
    }
    __syncthreads();

    // --- preload this wave's two A sub-tiles (32 rows x full C=256) ------
    const int rs0 = wave * 2, rs1 = wave * 2 + 1;
    v16bf aFrag0[8], aFrag1[8];
#pragma unroll
    for (int q = 0; q < 8; ++q) {
        aFrag0[q] = *(const v16bf*)(aT + (((rs0 * 8 + q) * 32) + lane) * 16);
        aFrag1[q] = *(const v16bf*)(aT + (((rs1 * 8 + q) * 32) + lane) * 16);
    }

    float vmin0[8], vmin1[8];
    int   vidx0[8], vidx1[8];
#pragma unroll
    for (int j = 0; j < 8; ++j) {
        vmin0[j] = 3.0e38f; vidx0[j] = 0;
        vmin1[j] = 3.0e38f; vidx1[j] = 0;
    }

    for (int kc = 0; kc < 8; ++kc) {           // 8 chunks of 128 codes
        // preload per-lane code norms for this chunk (overlaps the LDS copy)
        float ne[8];
#pragma unroll
        for (int t2 = 0; t2 < 8; ++t2)
            ne[t2] = wsNorm[kc * 128 + t2 * 16 + (lane & 15)];

        __syncthreads();                       // previous bT consumers done
        {   // copy 64KB pre-formatted fragment slab: global(L2) -> LDS
            const uint4* src = (const uint4*)(wsCb + (size_t)kc * 128 * CDIM);
            uint4* dst = (uint4*)bT;
#pragma unroll
            for (int i = 0; i < 32; ++i) dst[tid + i * 128] = src[tid + i * 128];
        }
        __syncthreads();

        if (kc < 7) {  // prefetch next slab into cache (global_prefetch_b8)
            const char* nxt = (const char*)(wsCb + (size_t)(kc + 1) * 128 * CDIM);
#pragma unroll
            for (int i = 0; i < 4; ++i)
                __builtin_prefetch(nxt + tid * 512 + i * 128, 0, 0);
        }

#pragma unroll 1
        for (int t2 = 0; t2 < 8; ++t2) {       // 16-code sub-tiles
            v8f acc0 = {}, acc1 = {};
#pragma unroll
            for (int q = 0; q < 8; ++q) {      // reduction over C in steps of 32
                v16bf bFrag = *(const v16bf*)(bT + (((t2 * 8 + q) * 32) + lane) * 16);
                acc0 = __builtin_amdgcn_wmma_f32_16x16x32_bf16(
                        false, aFrag0[q], false, bFrag, (short)0, acc0, false, false);
                acc1 = __builtin_amdgcn_wmma_f32_16x16x32_bf16(
                        false, aFrag1[q], false, bFrag, (short)0, acc1, false, false);
            }
            const int code = kc * 128 + t2 * 16 + (lane & 15);
            const float nt = ne[t2];
#pragma unroll
            for (int j = 0; j < 8; ++j) {
                float d0 = nt - 2.0f * acc0[j];    // ||z||^2 constant per row: dropped
                float d1 = nt - 2.0f * acc1[j];
                if (d0 < vmin0[j]) { vmin0[j] = d0; vidx0[j] = code; }
                if (d1 < vmin1[j]) { vmin1[j] = d1; vidx1[j] = code; }
            }
        }
    }

    // --- argmin across the 16 lanes of each half (wave32: masks<16 stay in half)
#pragma unroll
    for (int j = 0; j < 8; ++j) {
        float m0 = vmin0[j]; int i0 = vidx0[j];
        float m1 = vmin1[j]; int i1 = vidx1[j];
#pragma unroll
        for (int m = 1; m < 16; m <<= 1) {
            float o0 = __shfl_xor(m0, m, 32); int p0 = __shfl_xor(i0, m, 32);
            float o1 = __shfl_xor(m1, m, 32); int p1 = __shfl_xor(i1, m, 32);
            if (o0 < m0 || (o0 == m0 && p0 < i0)) { m0 = o0; i0 = p0; }
            if (o1 < m1 || (o1 == m1 && p1 < i1)) { m1 = o1; i1 = p1; }
        }
        vidx0[j] = i0; vidx1[j] = i1;
    }

    // C/D layout: slot j holds row j (lanes 0-15) and row j+8 (lanes 16-31).
    const int l15 = lane & 15;
    if (l15 < 8) {
        int mi0 = vidx0[0], mi1 = vidx1[0];
#pragma unroll
        for (int j = 1; j < 8; ++j) {
            if (l15 == j) { mi0 = vidx0[j]; mi1 = vidx1[j]; }
        }
        const int m = l15 + ((lane >> 4) << 3);
        const int row0 = rowBase + wave * 32 + m;        // sub-tile 0
        const int row1 = row0 + 16;                      // sub-tile 1
        wsIdx[row0]   = mi0;            wsIdx[row1]   = mi1;
        outIdxF[row0] = (float)mi0;     outIdxF[row1] = (float)mi1;
    }
}

// ---------------------------------------------------------------------------
// Kernel 2: gather value rows into [B,C,D,H,W] output + exact fp32 loss.
// Vectorized x4 over s: float4 z read / out write, int4 idx read, scalar
// codebook gathers from the 2MB L2-resident tables.
// ---------------------------------------------------------------------------
__global__ void vq_gather(const float* __restrict__ z,
                          const float* __restrict__ kw,
                          const float* __restrict__ vw,
                          const int* __restrict__ wsIdx,
                          float* __restrict__ out,
                          float* __restrict__ wsLoss) {
    const size_t g = ((size_t)blockIdx.x * 256 + threadIdx.x) * 4;  // element base
    const int bb  = (int)(g / ((size_t)CDIM * SVOL));
    const size_t rem = g - (size_t)bb * CDIM * SVOL;
    const int c = (int)(rem / SVOL);
    const int s = (int)(rem - (size_t)c * SVOL);        // multiple of 4

    const float4 zv = *(const float4*)(z + g);
    const int4   id = *(const int4*)(wsIdx + bb * SVOL + s);

    float4 ov;
    ov.x = vw[(size_t)id.x * CDIM + c];
    ov.y = vw[(size_t)id.y * CDIM + c];
    ov.z = vw[(size_t)id.z * CDIM + c];
    ov.w = vw[(size_t)id.w * CDIM + c];
    *(float4*)(out + g) = ov;

    float dx = zv.x - kw[(size_t)id.x * CDIM + c];
    float dy = zv.y - kw[(size_t)id.y * CDIM + c];
    float dz = zv.z - kw[(size_t)id.z * CDIM + c];
    float dw = zv.w - kw[(size_t)id.w * CDIM + c];
    float part = dx * dx + dy * dy + dz * dz + dw * dw;

    __shared__ float red[256];
    red[threadIdx.x] = part;
    __syncthreads();
    for (int st = 128; st > 0; st >>= 1) {
        if (threadIdx.x < st) red[threadIdx.x] += red[threadIdx.x + st];
        __syncthreads();
    }
    if (threadIdx.x == 0) atomicAdd(wsLoss, red[0]);
}

__global__ void vq_finalize(const float* __restrict__ wsLoss,
                            float* __restrict__ lossOut) {
    *lossOut = (1.0f + BETA) * (*wsLoss) / (float)((size_t)NROWS * CDIM);
}

// ---------------------------------------------------------------------------
extern "C" void kernel_launch(void* const* d_in, const int* in_sizes, int n_in,
                              void* d_out, int out_size, void* d_ws, size_t ws_size,
                              hipStream_t stream) {
    const float* z  = (const float*)d_in[0];
    const float* kw = (const float*)d_in[1];
    const float* vw = (const float*)d_in[2];

    float* out     = (float*)d_out;                       // z_q_value: 25,690,112 f32
    float* outIdx  = out + (size_t)2 * CDIM * SVOL;       // idx as f32: 100,352
    float* outLoss = outIdx + NROWS;                      // scalar loss

    // workspace: bf16 codebook fragments (512KB) | norms (4KB) | idx (392KB) | loss
    unsigned short* wsCb = (unsigned short*)d_ws;
    float* wsNorm = (float*)((char*)d_ws + (size_t)KCODES * CDIM * 2);
    int*   wsIdx  = (int*)((char*)wsNorm + (size_t)KCODES * 4);
    float* wsLoss = (float*)((char*)wsIdx + (size_t)NROWS * 4);

    vq_prep   <<<KCODES, 256, 0, stream>>>(kw, wsCb, wsNorm, wsLoss);
    vq_argmin <<<NROWS / 128, 128, 131072, stream>>>(z, wsCb, wsNorm, wsIdx, outIdx);
    vq_gather <<<(unsigned)((size_t)2 * CDIM * SVOL / 1024), 256, 0, stream>>>(
            z, kw, vw, wsIdx, out, wsLoss);
    vq_finalize<<<1, 1, 0, stream>>>(wsLoss, outLoss);
}
